// MultiQueryAttention_61813169324770
// MI455X (gfx1250) — compile-verified, matrix operand order checked by reading, not run on hardware
//
#include <hip/hip_runtime.h>
#include <hip/hip_bf16.h>

// ---------------------------------------------------------------------------
// MultiQueryAttention for MI455X (gfx1250), bf16 WMMA pipeline.
// B=8, S=1024, C=256, H=8, KD=256, softmax scale = 1/sqrt(32).
// ---------------------------------------------------------------------------

typedef __attribute__((ext_vector_type(16))) __bf16 v16bf;
typedef __attribute__((ext_vector_type(8)))  __bf16 v8bf;
typedef __attribute__((ext_vector_type(8)))  float  v8f;

#define BQ 8
#define SQ 1024
#define CQ 256
#define HQ 8
#define KDQ 256
// concat projection outputs: Wq rows [0,2048), Wk [2048,2304), Wv [2304,2560)
#define OPROJ 2560

union Frag16 { v16bf v; v8bf h[2]; };

__device__ inline v16bf load_frag16(const __bf16* p0, const __bf16* p1) {
  Frag16 u;
  u.h[0] = *(const v8bf*)p0;
  u.h[1] = *(const v8bf*)p1;
  return u.v;
}

__device__ inline v8f wmma_bf16(v16bf a, v16bf b, v8f c) {
  return __builtin_amdgcn_wmma_f32_16x16x32_bf16(
      /*neg_a=*/false, a, /*neg_b=*/false, b,
      /*c_mod=*/(short)0, c, /*reuse_a=*/false, /*reuse_b=*/false);
}

// Reduce across the 16 lanes of a half-wave (C/D fragment row lives in one half).
__device__ inline float half_max(float v) {
#pragma unroll
  for (int m = 1; m < 16; m <<= 1) v = fmaxf(v, __shfl_xor(v, m, 32));
  return v;
}
__device__ inline float half_sum(float v) {
#pragma unroll
  for (int m = 1; m < 16; m <<= 1) v += __shfl_xor(v, m, 32);
  return v;
}

// ---------------------------------------------------------------------------
// Kernel 0: f32 -> bf16 cast
// ---------------------------------------------------------------------------
__global__ void cast_f32_bf16_kernel(const float* __restrict__ src,
                                     __bf16* __restrict__ dst, int n) {
  int i = blockIdx.x * blockDim.x + threadIdx.x;
  if (i < n) dst[i] = (__bf16)src[i];
}

// ---------------------------------------------------------------------------
// Kernel 1: fused QKV projection.
//   P[token, o] = sum_c xb[token, c] * wcat[o, c]
// Wave computes a 16(M) x 64(N) tile. K = 256 is small enough to preload
// EVERY fragment (8 A + 32 B = 320 VGPRs) before any WMMA issues, so the
// 32 WMMAs drain with partial s_wait_loadcnt instead of stalling per step.
// __launch_bounds__(128,1): 4 waves = 1 wave/SIMD -> full VGPR budget.
// Scatter stores fold the torch raw-view quirks:
//   o < 2048          : qb[b][h=o/256][s][d=o%256]       ([B,H,S,KD])
//   2048 <= o < 2304  : kb[b][s][o-2048]                 ([B,S,KD])
//   o >= 2304         : vo=o-2304; i=vo*1024+s;
//                       vtb[b][d=i%256][t=i/256]         ([B,KD,S])
// grid: (BS/64, 2560/64), block 128 (4 waves, one M sub-tile each).
// ---------------------------------------------------------------------------
__global__ void __launch_bounds__(128, 1)
proj_qkv_kernel(const __bf16* __restrict__ xb,
                const __bf16* __restrict__ wcat,
                __bf16* __restrict__ qb,
                __bf16* __restrict__ kb,
                __bf16* __restrict__ vtb) {
  const int wave = threadIdx.x >> 5;
  const int lane = threadIdx.x & 31;
  const int n    = lane & 15;
  const int hf   = lane >> 4;

  const int m0 = blockIdx.x * 64 + wave * 16;
  const int n0 = blockIdx.y * 64;

  const __bf16* arow = xb + (size_t)(m0 + n) * CQ + 8 * hf;
  const __bf16* brow[4];
#pragma unroll
  for (int j = 0; j < 4; ++j)
    brow[j] = wcat + (size_t)(n0 + j * 16 + n) * CQ + 8 * hf;

  // ---- preload ALL fragments ------------------------------------------
  v16bf af[8];
#pragma unroll
  for (int kc = 0; kc < 8; ++kc)
    af[kc] = load_frag16(arow + kc * 32, arow + kc * 32 + 16);

  v16bf bf4[4][8];
#pragma unroll
  for (int j = 0; j < 4; ++j)
#pragma unroll
    for (int kc = 0; kc < 8; ++kc)
      bf4[j][kc] = load_frag16(brow[j] + kc * 32, brow[j] + kc * 32 + 16);

  // ---- 32 WMMAs --------------------------------------------------------
  v8f acc[4];
#pragma unroll
  for (int j = 0; j < 4; ++j)
    acc[j] = (v8f){0.f, 0.f, 0.f, 0.f, 0.f, 0.f, 0.f, 0.f};

#pragma unroll
  for (int kc = 0; kc < 8; ++kc)
#pragma unroll
    for (int j = 0; j < 4; ++j) acc[j] = wmma_bf16(af[kc], bf4[j][kc], acc[j]);

  // ---- scatter stores --------------------------------------------------
#pragma unroll
  for (int j = 0; j < 4; ++j) {
    const int o = n0 + j * 16 + n;  // uniform over r
    if (o < HQ * KDQ) {
      const int head = o >> 8, d = o & 255;
#pragma unroll
      for (int r = 0; r < 8; ++r) {
        const int token = m0 + r + 8 * hf;
        const int bb = token >> 10, s = token & 1023;
        qb[(((size_t)bb * HQ + head) * SQ + s) * KDQ + d] = (__bf16)acc[j][r];
      }
    } else if (o < HQ * KDQ + KDQ) {
      const int ko = o - HQ * KDQ;
#pragma unroll
      for (int r = 0; r < 8; ++r) {
        const int token = m0 + r + 8 * hf;
        const int bb = token >> 10, s = token & 1023;
        kb[((size_t)bb * SQ + s) * KDQ + ko] = (__bf16)acc[j][r];
      }
    } else {
      const int vo = o - (HQ * KDQ + KDQ);
#pragma unroll
      for (int r = 0; r < 8; ++r) {
        const int token = m0 + r + 8 * hf;
        const int bb = token >> 10, s = token & 1023;
        const int i = vo * 1024 + s;        // flat index in [KD,S] buffer
        const int d = i & 255, t = i >> 8;  // raw-view quirk: v[b,t,d]
        vtb[((size_t)bb * KDQ + d) * SQ + t] = (__bf16)acc[j][r];
      }
    }
  }
}

// ---------------------------------------------------------------------------
// Kernel 2: flash attention. grid (S/64, H, B), block 128 (4 waves).
// K/V tiles (32 keys) are cooperatively staged into double-buffered LDS so
// the 4 waves share one copy (4x VMEM reduction); one barrier per step.
// Wave owns 16 q-rows; online softmax; ctx accum in 128 f32 VGPRs.
// Writes ctx (bf16) scattered into a2 with the output raw-view quirk folded:
//   j = s*2048 + head*256 + d;  a2[b][j%1024][j/1024]   (a2: [B,1024,2048])
// ---------------------------------------------------------------------------
__global__ void __launch_bounds__(128, 1)
attn_kernel(const __bf16* __restrict__ qb,
            const __bf16* __restrict__ kb,
            const __bf16* __restrict__ vtb,
            __bf16* __restrict__ a2) {
  const int tid  = threadIdx.x;
  const int wave = tid >> 5;
  const int lane = tid & 31;
  const int n    = lane & 15;
  const int hf   = lane >> 4;

  const int s0   = blockIdx.x * 64 + wave * 16;
  const int head = blockIdx.y;
  const int bb   = blockIdx.z;

  __shared__ __align__(16) __bf16 kst[2][32][256];   // 2 x 16 KB
  __shared__ __align__(16) __bf16 vst[2][256][32];   // 2 x 16 KB
  __shared__ __align__(16) __bf16 pst[4][16][32];    // 4 KB (per-wave P tiles)

  const __bf16* kbb = kb  + (size_t)bb * SQ * KDQ;
  const __bf16* vbb = vtb + (size_t)bb * KDQ * SQ;

  // Cooperative staging of one 32-key K tile + V tile into LDS buffer `buf`.
  auto stage = [&](int buf, int t0) {
#pragma unroll
    for (int i = 0; i < 8; ++i) {                  // K: 1024 16B chunks / 128 thr
      const int chunk = tid + i * 128;
      const int tr = chunk >> 5;                   // key row 0..31
      const int tc = (chunk & 31) << 3;            // col 0..255 step 8
      *(v8bf*)&kst[buf][tr][tc] =
          *(const v8bf*)(kbb + (size_t)(t0 + tr) * KDQ + tc);
    }
#pragma unroll
    for (int i = 0; i < 8; ++i) {                  // V: 1024 16B chunks / 128 thr
      const int chunk = tid + i * 128;
      const int d  = chunk >> 2;                   // dim row 0..255
      const int tc = (chunk & 3) << 3;             // key col 0,8,16,24
      *(v8bf*)&vst[buf][d][tc] =
          *(const v8bf*)(vbb + (size_t)d * SQ + t0 + tc);
    }
  };

  // Q fragments for this wave's 16 rows (lane m = n), all 8 K-chunks.
  const __bf16* qrow = qb + (((size_t)bb * HQ + head) * SQ + (s0 + n)) * KDQ;
  v16bf aq[8];
#pragma unroll
  for (int kc = 0; kc < 8; ++kc)
    aq[kc] = load_frag16(qrow + kc * 32 + 8 * hf, qrow + kc * 32 + 16 + 8 * hf);

  v8f Oacc[16];
#pragma unroll
  for (int dt = 0; dt < 16; ++dt)
    Oacc[dt] = (v8f){0.f, 0.f, 0.f, 0.f, 0.f, 0.f, 0.f, 0.f};

  float mrow[8], lrow[8];
#pragma unroll
  for (int r = 0; r < 8; ++r) { mrow[r] = -3.0e38f; lrow[r] = 0.f; }

  const float scale = 0.17677669529663687f;  // 1/sqrt(32)

  stage(0, 0);

  for (int it = 0; it < SQ / 32; ++it) {
    const int cur = it & 1;
    __syncthreads();                     // staging of `cur` complete everywhere
    if (it + 1 < SQ / 32) stage(cur ^ 1, (it + 1) * 32);  // overlap next tile

    // ---- scores: 16x32 tile as two 16x16 fragments (K from LDS) -------
    v8f sc0 = {0.f, 0.f, 0.f, 0.f, 0.f, 0.f, 0.f, 0.f};
    v8f sc1 = sc0;
#pragma unroll
    for (int kc = 0; kc < 8; ++kc) {
      const __bf16* kr0 = &kst[cur][n][kc * 32];
      const __bf16* kr1 = &kst[cur][16 + n][kc * 32];
      v16bf b0 = load_frag16(kr0 + 8 * hf, kr0 + 16 + 8 * hf);
      v16bf b1 = load_frag16(kr1 + 8 * hf, kr1 + 16 + 8 * hf);
      sc0 = wmma_bf16(aq[kc], b0, sc0);
      sc1 = wmma_bf16(aq[kc], b1, sc1);
    }

    // ---- online softmax update ---------------------------------------
    float cfa[8];
#pragma unroll
    for (int r = 0; r < 8; ++r) {
      float a0 = sc0[r] * scale;
      float a1 = sc1[r] * scale;
      float rm = half_max(fmaxf(a0, a1));
      float mn = fmaxf(mrow[r], rm);
      float cf = __expf(mrow[r] - mn);
      mrow[r] = mn;
      a0 = __expf(a0 - mn);
      a1 = __expf(a1 - mn);
      sc0[r] = a0;
      sc1[r] = a1;
      lrow[r] = lrow[r] * cf + half_sum(a0 + a1);
      cfa[r] = cf;
    }
#pragma unroll
    for (int dt = 0; dt < 16; ++dt)
#pragma unroll
      for (int r = 0; r < 8; ++r) Oacc[dt][r] *= cfa[r];

    // ---- C-layout P -> A-layout bf16 fragment via per-wave LDS tile ---
#pragma unroll
    for (int r = 0; r < 8; ++r) {
      pst[wave][r + 8 * hf][n]      = (__bf16)sc0[r];
      pst[wave][r + 8 * hf][16 + n] = (__bf16)sc1[r];
    }
    asm volatile("s_wait_dscnt 0" ::: "memory");  // DS in-order in-wave + fence
    v16bf ap = load_frag16(&pst[wave][n][8 * hf], &pst[wave][n][16 + 8 * hf]);
    asm volatile("s_wait_dscnt 0" ::: "memory");

    // ---- ctx += P @ V  (V from LDS, [KD][32]: contiguous B-frag loads)
#pragma unroll
    for (int dt = 0; dt < 16; ++dt) {
      const __bf16* vr = &vst[cur][dt * 16 + n][0];
      v16bf bv = load_frag16(vr + 8 * hf, vr + 16 + 8 * hf);
      Oacc[dt] = wmma_bf16(ap, bv, Oacc[dt]);
    }
  }

  // ---- finalize + scatter (output raw-view quirk folded) -------------
  float linv[8];
#pragma unroll
  for (int r = 0; r < 8; ++r) linv[r] = 1.0f / lrow[r];
#pragma unroll
  for (int dt = 0; dt < 16; ++dt) {
#pragma unroll
    for (int r = 0; r < 8; ++r) {
      const int m = r + 8 * hf;
      const int s = s0 + m;
      const int d = dt * 16 + n;
      const int j = s * (HQ * KDQ) + head * KDQ + d;  // flat [S,H*KD] index
      a2[((size_t)bb * 1024 + (j & 1023)) * 2048 + (j >> 10)] =
          (__bf16)(Oacc[dt][r] * linv[r]);
    }
  }
}

// ---------------------------------------------------------------------------
// Kernel 3: output GEMM.
//   out[b, p, o] = sum_{c<2048} a2[b][p][c] * Wo[o][c]   (f32 out, [B,S,C])
// Wave computes a 16(M) x 64(N) tile, K = 2048 processed in groups of 4
// k-steps with register double buffering: the next group's 20 fragments
// (160 VGPRs) load while the current group's 16 WMMAs execute.
// __launch_bounds__(128,1): 1 wave/SIMD -> full VGPR budget for the pipeline.
// grid (BS/64, 256/64), block 128 (4 waves, one M sub-tile each).
// ---------------------------------------------------------------------------
__global__ void __launch_bounds__(128, 1)
out_gemm_kernel(const __bf16* __restrict__ a2,
                const __bf16* __restrict__ wob,
                float* __restrict__ out) {
  const int wave = threadIdx.x >> 5;
  const int lane = threadIdx.x & 31;
  const int n    = lane & 15;
  const int hf   = lane >> 4;

  const int m0 = blockIdx.x * 64 + wave * 16;
  const int o0 = blockIdx.y * 64;
  const int bb = m0 >> 10;
  const int p0 = m0 & 1023;

  const __bf16* arow = a2 + ((size_t)bb * 1024 + p0 + n) * 2048 + 8 * hf;
  const __bf16* brow[4];
#pragma unroll
  for (int j = 0; j < 4; ++j)
    brow[j] = wob + (size_t)(o0 + j * 16 + n) * 2048 + 8 * hf;

  v8f acc[4];
#pragma unroll
  for (int j = 0; j < 4; ++j)
    acc[j] = (v8f){0.f, 0.f, 0.f, 0.f, 0.f, 0.f, 0.f, 0.f};

  // Double-buffered fragment groups: 4 k-steps per group, 16 groups total.
  v16bf aG[2][4];
  v16bf bG[2][4][4];

  auto load_group = [&](int buf, int g) {
#pragma unroll
    for (int i = 0; i < 4; ++i) {
      const int off = (g * 4 + i) * 32;
      aG[buf][i] = load_frag16(arow + off, arow + off + 16);
#pragma unroll
      for (int j = 0; j < 4; ++j)
        bG[buf][i][j] = load_frag16(brow[j] + off, brow[j] + off + 16);
    }
  };

  load_group(0, 0);
#pragma unroll
  for (int g = 0; g < 16; ++g) {
    const int cur = g & 1;
    if (g + 1 < 16) load_group(cur ^ 1, g + 1);
#pragma unroll
    for (int i = 0; i < 4; ++i)
#pragma unroll
      for (int j = 0; j < 4; ++j)
        acc[j] = wmma_bf16(aG[cur][i], bG[cur][i][j], acc[j]);
  }

#pragma unroll
  for (int j = 0; j < 4; ++j) {
#pragma unroll
    for (int r = 0; r < 8; ++r) {
      const int m = r + 8 * hf;
      out[((size_t)bb * SQ + p0 + m) * CQ + o0 + j * 16 + n] = acc[j][r];
    }
  }
}

// ---------------------------------------------------------------------------
// Host launcher
// ---------------------------------------------------------------------------
extern "C" void kernel_launch(void* const* d_in, const int* in_sizes, int n_in,
                              void* d_out, int out_size, void* d_ws, size_t ws_size,
                              hipStream_t stream) {
  (void)in_sizes; (void)n_in; (void)out_size; (void)ws_size;

  const float* x  = (const float*)d_in[0];  // [B,S,C]
  const float* Wq = (const float*)d_in[1];  // [2048,256]
  const float* Wk = (const float*)d_in[2];  // [256,256]
  const float* Wv = (const float*)d_in[3];  // [256,256]
  const float* Wo = (const float*)d_in[4];  // [256,2048]
  float* out = (float*)d_out;               // [B,S,C]

  char* ws = (char*)d_ws;
  size_t off = 0;
  auto alloc = [&](size_t bytes) { char* p = ws + off; off += (bytes + 255) & ~size_t(255); return p; };

  __bf16* xb   = (__bf16*)alloc((size_t)BQ * SQ * CQ * 2);          // 4 MB
  __bf16* wcat = (__bf16*)alloc((size_t)OPROJ * CQ * 2);            // 1.25 MB
  __bf16* wob  = (__bf16*)alloc((size_t)CQ * HQ * KDQ * 2);         // 1 MB
  __bf16* qb   = (__bf16*)alloc((size_t)BQ * HQ * SQ * KDQ * 2);    // 32 MB
  __bf16* kb   = (__bf16*)alloc((size_t)BQ * SQ * KDQ * 2);         // 4 MB
  __bf16* vtb  = (__bf16*)alloc((size_t)BQ * KDQ * SQ * 2);         // 4 MB
  __bf16* a2   = (__bf16*)alloc((size_t)BQ * 1024 * 2048 * 2);      // 32 MB

  auto cast = [&](const float* s, __bf16* d, int n) {
    cast_f32_bf16_kernel<<<(n + 255) / 256, 256, 0, stream>>>(s, d, n);
  };
  cast(x,  xb,                 BQ * SQ * CQ);
  cast(Wq, wcat,               HQ * KDQ * CQ);          // rows [0,2048)
  cast(Wk, wcat + 2048 * CQ,   KDQ * CQ);               // rows [2048,2304)
  cast(Wv, wcat + 2304 * CQ,   KDQ * CQ);               // rows [2304,2560)
  cast(Wo, wob,                CQ * HQ * KDQ);

  {
    dim3 grid(BQ * SQ / 64, OPROJ / 64);
    proj_qkv_kernel<<<grid, 128, 0, stream>>>(xb, wcat, qb, kb, vtb);
  }
  {
    dim3 grid(SQ / 64, HQ, BQ);
    attn_kernel<<<grid, 128, 0, stream>>>(qb, kb, vtb, a2);
  }
  {
    dim3 grid(BQ * SQ / 64, CQ / 64);
    out_gemm_kernel<<<grid, 128, 0, stream>>>(a2, wob, out);
  }
}
